// Spikformer_32873679683858
// MI455X (gfx1250) — 16-bit and 8-bit matrix operand order repaired and checked by reading, compiled, not run on hardware
//
#include <hip/hip_runtime.h>

typedef __attribute__((ext_vector_type(16))) _Float16 v16h;
typedef __attribute__((ext_vector_type(8)))  float    v8f;
typedef __attribute__((ext_vector_type(8)))  int      v8i;
typedef __attribute__((ext_vector_type(4)))  unsigned int v4u;
typedef __attribute__((ext_vector_type(2)))  unsigned int v2u;

namespace {
constexpr int TT = 4, BB = 2, NN = 1024, CC = 384, HH = 12, HID = 1536;
constexpr int MM = TT * BB * NN;
}

union FragH { v16h v; v4u q[2]; };
union Frag8 { v8i v; v4u q[2]; v2u d[4]; unsigned int w[8]; };

#define CDIV(a, b) (((a) + (b) - 1) / (b))

__global__ void k_transpose_cast(const float* __restrict__ W, _Float16* __restrict__ Wt,
                                 int din, int dout) {
  int idx = blockIdx.x * blockDim.x + threadIdx.x;
  if (idx >= din * dout) return;
  int o = idx % dout, i = idx / dout;
  Wt[(size_t)o * din + i] = (_Float16)W[(size_t)i * dout + o];
}

__global__ void k_fold_bn(const float* __restrict__ b, const float* __restrict__ g,
                          const float* __restrict__ be, const float* __restrict__ rm,
                          const float* __restrict__ rv, float* __restrict__ sc,
                          float* __restrict__ bi, int dout) {
  int i = blockIdx.x * blockDim.x + threadIdx.x;
  if (i >= dout) return;
  float s = g[i] * rsqrtf(rv[i] + 1e-5f);
  sc[i] = s;
  bi[i] = (b[i] - rm[i]) * s + be[i];
}

__global__ void k_cast_f16(const float* __restrict__ x, _Float16* __restrict__ y, int n) {
  int i = blockIdx.x * blockDim.x + threadIdx.x;
  if (i < n) y[i] = (_Float16)x[i];
}

__global__ void k_gemm_bn_f16(const _Float16* __restrict__ A, const _Float16* __restrict__ Wt,
                              const float* __restrict__ sc, const float* __restrict__ bi,
                              float* __restrict__ U, int M, int K, int Nout) {
  const int lane = threadIdx.x & 31;
  const int wave = (blockIdx.x * blockDim.x + threadIdx.x) >> 5;
  const int tN = Nout >> 5;
  const int tm = wave / tN, tn = wave % tN;
  if (tm * 32 >= M) return;
  const int r = lane & 15, hh = lane >> 4;
  const int m0 = tm * 32, n0 = tn * 32;

  v8f acc[2][2] = {};
  for (int kb = 0; kb < K; kb += 32) {
    FragH a[2], b[2];
#pragma unroll
    for (int i = 0; i < 2; ++i) {
      const _Float16* pa = A + (size_t)(m0 + 16 * i + r) * K + kb + hh * 8;
      a[i].q[0] = *(const v4u*)pa;
      a[i].q[1] = *(const v4u*)(pa + 16);
    }
#pragma unroll
    for (int j = 0; j < 2; ++j) {
      const _Float16* pb = Wt + (size_t)(n0 + 16 * j + r) * K + kb + hh * 8;
      b[j].q[0] = *(const v4u*)pb;
      b[j].q[1] = *(const v4u*)(pb + 16);
    }
#pragma unroll
    for (int i = 0; i < 2; ++i)
#pragma unroll
      for (int j = 0; j < 2; ++j)
        acc[i][j] = __builtin_amdgcn_wmma_f32_16x16x32_f16(
            false, a[i].v, false, b[j].v, (short)0, acc[i][j], false, false);
  }

#pragma unroll
  for (int i = 0; i < 2; ++i)
#pragma unroll
    for (int j = 0; j < 2; ++j) {
      int col = n0 + 16 * j + r;
      float s = sc[col], t = bi[col];
#pragma unroll
      for (int g = 0; g < 8; ++g) {
        int row = m0 + 16 * i + (hh ? g + 8 : g);
        U[(size_t)row * Nout + col] = acc[i][j][g] * s + t;
      }
    }
}


__global__ void k_lif_qk_u8(const float* __restrict__ U, const float* __restrict__ v0,
                            unsigned char* __restrict__ S) {
  int idx = blockIdx.x * blockDim.x + threadIdx.x;
  if (idx >= BB * NN * CC) return;
  int c = idx % CC, n = (idx / CC) % NN, b = idx / (CC * NN);
  int h = c >> 5, d = c & 31;
  float v = v0[(size_t)n * CC + c];
#pragma unroll
  for (int t = 0; t < TT; ++t) {
    float x = U[(size_t)((t * BB + b) * NN + n) * CC + c];
    v = 0.5f * (v + x);
    float s = (v >= 1.0f) ? 1.0f : 0.0f;
    S[(size_t)(((t * BB + b) * HH + h) * NN + n) * 32 + d] = (unsigned char)s;
    v *= (1.0f - s);
  }
}

__global__ void k_lif_v_u8t(const float* __restrict__ U, const float* __restrict__ v0,
                            unsigned char* __restrict__ S) {
  int idx = blockIdx.x * blockDim.x + threadIdx.x;
  if (idx >= BB * NN * CC) return;
  int c = idx % CC, n = (idx / CC) % NN, b = idx / (CC * NN);
  int h = c >> 5, d = c & 31;
  float v = v0[(size_t)n * CC + c];
#pragma unroll
  for (int t = 0; t < TT; ++t) {
    float x = U[(size_t)((t * BB + b) * NN + n) * CC + c];
    v = 0.5f * (v + x);
    float s = (v >= 1.0f) ? 1.0f : 0.0f;
    S[(size_t)(((t * BB + b) * HH + h) * 32 + d) * NN + n] = (unsigned char)s;
    v *= (1.0f - s);
  }
}

__global__ void k_lif_f16(const float* __restrict__ U, const float* __restrict__ v0,
                          _Float16* __restrict__ S, int D, float vth) {
  int idx = blockIdx.x * blockDim.x + threadIdx.x;
  if (idx >= BB * NN * D) return;
  int d = idx % D, n = (idx / D) % NN, b = idx / (D * NN);
  float v = v0[(size_t)n * D + d];
#pragma unroll
  for (int t = 0; t < TT; ++t) {
    float x = U[(size_t)((t * BB + b) * NN + n) * D + d];
    v = 0.5f * (v + x);
    float s = (v >= vth) ? 1.0f : 0.0f;
    S[(size_t)((t * BB + b) * NN + n) * D + d] = (_Float16)s;
    v *= (1.0f - s);
  }
}

__global__ void k_lif_add(const float* __restrict__ U, const float* __restrict__ v0,
                          const float* __restrict__ base, float* __restrict__ out,
                          _Float16* __restrict__ out_h) {
  int idx = blockIdx.x * blockDim.x + threadIdx.x;
  if (idx >= BB * NN * CC) return;
  int c = idx % CC, n = (idx / CC) % NN, b = idx / (CC * NN);
  float v = v0[(size_t)n * CC + c];
#pragma unroll
  for (int t = 0; t < TT; ++t) {
    size_t off = (size_t)((t * BB + b) * NN + n) * CC + c;
    float x = U[off];
    v = 0.5f * (v + x);
    float s = (v >= 1.0f) ? 1.0f : 0.0f;
    float y = base[off] + s;
    out[off] = y;
    if (out_h) out_h[off] = (_Float16)y;
    v *= (1.0f - s);
  }
}

__global__ void k_attn_iu8(const unsigned char* __restrict__ Q,
                           const unsigned char* __restrict__ Kk,
                           const unsigned char* __restrict__ Vt,
                           float* __restrict__ O) {
  __shared__ __align__(16) unsigned char lds[8][16 * 64];
  const int lane = threadIdx.x & 31;
  const int wloc = threadIdx.x >> 5;
  const int wave = (blockIdx.x * blockDim.x + threadIdx.x) >> 5;
  int nt = wave & 63;
  int rest = wave >> 6;
  int h = rest % HH; rest /= HH;
  int b = rest % BB;
  int t = rest / BB;
  const int r = lane & 15, hh = lane >> 4;

  const size_t head = (size_t)((t * BB + b) * HH + h) * NN * 32;
  const unsigned char* q = Q + head;
  const unsigned char* k = Kk + head;
  const unsigned char* v = Vt + head;

  Frag8 aq;
  {
    const unsigned char* p = q + (size_t)(nt * 16 + r) * 32 + hh * 8;
    aq.d[0] = *(const v2u*)p;
    aq.d[1] = *(const v2u*)(p + 16);
    aq.w[4] = aq.w[5] = aq.w[6] = aq.w[7] = 0;
  }

  v8i oacc[2] = {};
  unsigned char* myl = lds[wloc];

  for (int mb = 0; mb < NN; mb += 64) {
#pragma unroll
    for (int mt = 0; mt < 4; ++mt) {
      Frag8 bk;
      const unsigned char* p = k + (size_t)(mb + mt * 16 + r) * 32 + hh * 8;
      bk.d[0] = *(const v2u*)p;
      bk.d[1] = *(const v2u*)(p + 16);
      bk.w[4] = bk.w[5] = bk.w[6] = bk.w[7] = 0;
      v8i c0 = {};
      v8i att = __builtin_amdgcn_wmma_i32_16x16x64_iu8(false, aq.v, false, bk.v,
                                                       c0, false, false);
#pragma unroll
      for (int g = 0; g < 8; ++g) {
        int row = hh ? g + 8 : g;
        myl[row * 64 + mt * 16 + r] = (unsigned char)att[g];
      }
    }
    __syncthreads();
    Frag8 aa;
    {
      const unsigned char* p = myl + r * 64 + hh * 16;
      aa.q[0] = *(const v4u*)p;
      aa.q[1] = *(const v4u*)(p + 32);
    }
#pragma unroll
    for (int j = 0; j < 2; ++j) {
      Frag8 bv;
      const unsigned char* p = v + (size_t)(j * 16 + r) * NN + mb + hh * 16;
      bv.q[0] = *(const v4u*)p;
      bv.q[1] = *(const v4u*)(p + 32);
      oacc[j] = __builtin_amdgcn_wmma_i32_16x16x64_iu8(false, aa.v, false, bv.v,
                                                       oacc[j], false, false);
    }
    __syncthreads();
  }

  const size_t ob = (size_t)(t * BB + b) * NN * CC + (size_t)h * 32;
#pragma unroll
  for (int j = 0; j < 2; ++j) {
    int dd = j * 16 + r;
#pragma unroll
    for (int g = 0; g < 8; ++g) {
      int n = nt * 16 + (hh ? g + 8 : g);
      O[ob + (size_t)n * CC + dd] = (float)oacc[j][g] * 0.125f;
    }
  }
}

extern "C" void kernel_launch(void* const* d_in, const int* in_sizes, int n_in,
                              void* d_out, int out_size, void* d_ws, size_t ws_size,
                              hipStream_t stream) {
  (void)in_sizes; (void)n_in; (void)out_size; (void)ws_size;
  const float* x = (const float*)d_in[0];
  auto P = [&](int i) { return (const float*)d_in[i]; };
  const float *Wq = P(1),  *v0q = P(7);
  const float *Wk = P(8),  *v0k = P(14);
  const float *Wv = P(15), *v0v = P(21);
  const float *Wp = P(22), *v0p = P(28);
  const float *av0 = P(29);
  const float *W1 = P(30), *v01 = P(36);
  const float *W2 = P(37), *v02 = P(43);

  char* cur = (char*)d_ws;
  auto alloc = [&](size_t bytes) -> void* {
    void* p = cur;
    cur += (bytes + 255) & ~(size_t)255;
    return p;
  };
  _Float16* Xh   = (_Float16*)alloc((size_t)MM * CC * 2);
  _Float16* Wtq  = (_Float16*)alloc((size_t)CC * CC * 2);
  _Float16* Wtk  = (_Float16*)alloc((size_t)CC * CC * 2);
  _Float16* Wtv  = (_Float16*)alloc((size_t)CC * CC * 2);
  _Float16* Wtp  = (_Float16*)alloc((size_t)CC * CC * 2);
  _Float16* Wt1  = (_Float16*)alloc((size_t)CC * HID * 2);
  _Float16* Wt2  = (_Float16*)alloc((size_t)HID * CC * 2);
  float* sq = (float*)alloc(CC * 4); float* tq = (float*)alloc(CC * 4);
  float* sk = (float*)alloc(CC * 4); float* tk = (float*)alloc(CC * 4);
  float* sv = (float*)alloc(CC * 4); float* tv = (float*)alloc(CC * 4);
  float* sp = (float*)alloc(CC * 4); float* tp = (float*)alloc(CC * 4);
  float* s1 = (float*)alloc(HID * 4); float* t1 = (float*)alloc(HID * 4);
  float* s2 = (float*)alloc(CC * 4); float* t2 = (float*)alloc(CC * 4);
  float* Uq = (float*)alloc((size_t)MM * CC * 4);
  float* Uk = (float*)alloc((size_t)MM * CC * 4);
  float* Uv = (float*)alloc((size_t)MM * CC * 4);
  unsigned char* qs = (unsigned char*)alloc((size_t)MM * CC);
  unsigned char* ks = (unsigned char*)alloc((size_t)MM * CC);
  unsigned char* vt = (unsigned char*)alloc((size_t)MM * CC);
  _Float16* ospk = (_Float16*)alloc((size_t)MM * CC * 2);
  float* U1 = (float*)alloc((size_t)MM * HID * 4);
  _Float16* hspk = (_Float16*)alloc((size_t)MM * HID * 2);
  float* O  = Uq;
  float* Up = Uk;
  float* x1 = Uv;
  _Float16* x1h = Xh;
  float* U2 = O;

  const int TPB = 256;
  k_transpose_cast<<<CDIV(CC * CC, TPB), TPB, 0, stream>>>(Wq, Wtq, CC, CC);
  k_transpose_cast<<<CDIV(CC * CC, TPB), TPB, 0, stream>>>(Wk, Wtk, CC, CC);
  k_transpose_cast<<<CDIV(CC * CC, TPB), TPB, 0, stream>>>(Wv, Wtv, CC, CC);
  k_transpose_cast<<<CDIV(CC * CC, TPB), TPB, 0, stream>>>(Wp, Wtp, CC, CC);
  k_transpose_cast<<<CDIV(CC * HID, TPB), TPB, 0, stream>>>(W1, Wt1, CC, HID);
  k_transpose_cast<<<CDIV(HID * CC, TPB), TPB, 0, stream>>>(W2, Wt2, HID, CC);
  k_fold_bn<<<CDIV(CC, TPB), TPB, 0, stream>>>(P(2), P(3), P(4), P(5), P(6), sq, tq, CC);
  k_fold_bn<<<CDIV(CC, TPB), TPB, 0, stream>>>(P(9), P(10), P(11), P(12), P(13), sk, tk, CC);
  k_fold_bn<<<CDIV(CC, TPB), TPB, 0, stream>>>(P(16), P(17), P(18), P(19), P(20), sv, tv, CC);
  k_fold_bn<<<CDIV(CC, TPB), TPB, 0, stream>>>(P(23), P(24), P(25), P(26), P(27), sp, tp, CC);
  k_fold_bn<<<CDIV(HID, TPB), TPB, 0, stream>>>(P(31), P(32), P(33), P(34), P(35), s1, t1, HID);
  k_fold_bn<<<CDIV(CC, TPB), TPB, 0, stream>>>(P(38), P(39), P(40), P(41), P(42), s2, t2, CC);
  k_cast_f16<<<CDIV(MM * CC, TPB), TPB, 0, stream>>>(x, Xh, MM * CC);

  auto gemm_blocks = [&](int M, int Nout) { return (M / 32) * (Nout / 32) * 32 / TPB; };
  k_gemm_bn_f16<<<gemm_blocks(MM, CC), TPB, 0, stream>>>(Xh, Wtq, sq, tq, Uq, MM, CC, CC);
  k_gemm_bn_f16<<<gemm_blocks(MM, CC), TPB, 0, stream>>>(Xh, Wtk, sk, tk, Uk, MM, CC, CC);
  k_gemm_bn_f16<<<gemm_blocks(MM, CC), TPB, 0, stream>>>(Xh, Wtv, sv, tv, Uv, MM, CC, CC);
  k_lif_qk_u8<<<CDIV(BB * NN * CC, TPB), TPB, 0, stream>>>(Uq, v0q, qs);
  k_lif_qk_u8<<<CDIV(BB * NN * CC, TPB), TPB, 0, stream>>>(Uk, v0k, ks);
  k_lif_v_u8t<<<CDIV(BB * NN * CC, TPB), TPB, 0, stream>>>(Uv, v0v, vt);
  k_attn_iu8<<<TT * BB * HH * (NN / 16) * 32 / TPB, TPB, 0, stream>>>(qs, ks, vt, O);
  k_lif_f16<<<CDIV(BB * NN * CC, TPB), TPB, 0, stream>>>(O, av0, ospk, CC, 0.5f);
  k_gemm_bn_f16<<<gemm_blocks(MM, CC), TPB, 0, stream>>>(ospk, Wtp, sp, tp, Up, MM, CC, CC);
  k_lif_add<<<CDIV(BB * NN * CC, TPB), TPB, 0, stream>>>(Up, v0p, x, x1, x1h);
  k_gemm_bn_f16<<<gemm_blocks(MM, HID), TPB, 0, stream>>>(x1h, Wt1, s1, t1, U1, MM, CC, HID);
  k_lif_f16<<<CDIV(BB * NN * HID, TPB), TPB, 0, stream>>>(U1, v01, hspk, HID, 1.0f);
  k_gemm_bn_f16<<<gemm_blocks(MM, CC), TPB, 0, stream>>>(hspk, Wt2, s2, t2, U2, MM, HID, CC);
  k_lif_add<<<CDIV(BB * NN * CC, TPB), TPB, 0, stream>>>(U2, v02, x1, (float*)d_out,
                                                        (_Float16*)nullptr);
}
